// MinkowskiResBlock_38972533244056
// MI455X (gfx1250) — matrix operand order repaired and checked by reading, not compiled, not run on hardware
//
#include <hip/hip_runtime.h>

#define NPTS 131072
#define CCH  192
#define KOFF 27
#define NT16 12          // 192 / 16 output-column tiles
#define KC32 6           // 192 / 32 K chunks
#define LDSS 208         // padded LDS row stride in bf16 elems (416B, 16B aligned)
#define MROWS 32         // rows per workgroup (2 row-tiles of 16)

typedef __bf16 v16bf __attribute__((ext_vector_type(16)));
typedef float  v8f   __attribute__((ext_vector_type(8)));

__device__ __forceinline__ unsigned short f2bf(float f) {
    union { float f; unsigned u; } x; x.f = f;
    unsigned r = x.u + 0x7FFFu + ((x.u >> 16) & 1u);   // round-to-nearest-even
    return (unsigned short)(r >> 16);
}

// ---- feats f32 -> bf16 with zero pad row at index NPTS ----
__global__ void mk_cvt_feats(const float* __restrict__ f, unsigned short* __restrict__ o) {
    long i = (long)blockIdx.x * blockDim.x + threadIdx.x;
    const long total = (long)(NPTS + 1) * CCH;
    if (i >= total) return;
    o[i] = (i < (long)NPTS * CCH) ? f2bf(f[i]) : (unsigned short)0;
}

// ---- W [K,C,C] f32 -> packed bf16 B-fragments ----
// packed elem index: (((k*NT16 + nt)*KC32 + kc)*32 + lane)*16 + j
// lane<16 : column nt*16+lane,      K = kc*32 + j       (j=0..15)
// lane>=16: column nt*16+(lane-16), K = kc*32 + 16 + j
__global__ void mk_pack_w(const float* __restrict__ W, unsigned short* __restrict__ P) {
    long i = (long)blockIdx.x * blockDim.x + threadIdx.x;
    const long total = (long)KOFF * CCH * CCH;
    if (i >= total) return;
    int j    = (int)(i & 15);
    int lane = (int)((i >> 4) & 31);
    long r   = i >> 9;
    int kc   = (int)(r % KC32); r /= KC32;
    int nt   = (int)(r % NT16); r /= NT16;
    int k    = (int)r;
    int cin  = kc * 32 + ((lane >> 4) << 4) + j;
    int cout = nt * 16 + (lane & 15);
    P[i] = f2bf(W[((long)k * CCH + cin) * CCH + cout]);
}

// ---- gather-GEMM: out[N,C] = sum_k src[neigh[k]] @ Wk ----
// 128 threads = 4 waves; WG covers 32 rows; wave owns 3 N-tiles x 2 row-tiles.
// Double-buffered LDS A tiles: stage k+1 while computing k.
__global__ __launch_bounds__(128)
void mk_spconv(const unsigned short* __restrict__ src,   // [NPTS+1, CCH] bf16
               const unsigned short* __restrict__ wp,    // packed bf16 weights
               const int*            __restrict__ neigh, // [KOFF, NPTS]
               float*                __restrict__ out)   // [NPTS, CCH] f32
{
    __shared__ __align__(16) unsigned short At[2][MROWS * LDSS];
    const int tid  = threadIdx.x;
    const int lane = tid & 31;
    const int wave = tid >> 5;      // 0..3 -> N-tiles wave*3 .. wave*3+2
    const int half = lane >> 4;
    const int mrow = lane & 15;
    const int rt   = blockIdx.x;    // 32-row tile index
    const int srow = tid >> 2;      // staging: row 0..31 (4 threads per row)
    const int ssub = tid & 3;       // staging: 6 x b128 chunks each

    v8f a00 = {}, a01 = {}, a02 = {};   // row-tile 0, N-tiles t=0..2
    v8f a10 = {}, a11 = {}, a12 = {};   // row-tile 1

    // ---- stage tile k=0 into buffer 0 ----
    {
        int idx = neigh[(long)0 * NPTS + rt * MROWS + srow];
        const uint4* srcRow = (const uint4*)(src + (long)idx * CCH);
        uint4*       dstRow = (uint4*)(At[0] + srow * LDSS);
        #pragma unroll
        for (int s = 0; s < 6; ++s) {
            int c128 = ssub * 6 + s;            // 0..23 (24 x 16B = 384B row)
            dstRow[c128] = srcRow[c128];
        }
    }

    for (int k = 0; k < KOFF; ++k) {
        __syncthreads();            // buf[k&1] staged; prior reads of buf[1-(k&1)] done
        const int cur = k & 1;

        // ---- prefetch-stage tile k+1 into the other buffer ----
        if (k + 1 < KOFF) {
            int idx = neigh[(long)(k + 1) * NPTS + rt * MROWS + srow];
            const uint4* srcRow = (const uint4*)(src + (long)idx * CCH);
            uint4*       dstRow = (uint4*)(At[cur ^ 1] + srow * LDSS);
            #pragma unroll
            for (int s = 0; s < 6; ++s) {
                int c128 = ssub * 6 + s;
                dstRow[c128] = srcRow[c128];
            }
        }

        // ---- compute on buf[cur] ----
        const unsigned short* wk = wp + (long)k * (NT16 * KC32) * 512; // 512 ushort/frag
        #pragma unroll
        for (int kc = 0; kc < KC32; ++kc) {
            // A fragments (contiguous K, matching packed B): low lanes K kb+0..15,
            // high lanes K kb+16..31
            union { uint4 q[2]; v16bf v; } fa0, fa1;
            const uint4* ap0 =
                (const uint4*)(At[cur] + (0 * 16 + mrow) * LDSS + kc * 32 + half * 16);
            const uint4* ap1 =
                (const uint4*)(At[cur] + (1 * 16 + mrow) * LDSS + kc * 32 + half * 16);
            fa0.q[0] = ap0[0]; fa0.q[1] = ap0[1];
            fa1.q[0] = ap1[0]; fa1.q[1] = ap1[1];
            #pragma unroll
            for (int t = 0; t < 3; ++t) {
                int nt = wave * 3 + t;
                const uint4* bp =
                    (const uint4*)(wk + ((long)(nt * KC32 + kc) * 32 + lane) * 16);
                union { uint4 q[2]; v16bf v; } fb;
                fb.q[0] = bp[0];
                fb.q[1] = bp[1];
                v8f& r0 = (t == 0) ? a00 : ((t == 1) ? a01 : a02);
                v8f& r1 = (t == 0) ? a10 : ((t == 1) ? a11 : a12);
                r0 = __builtin_amdgcn_wmma_f32_16x16x32_bf16(
                         false, fa0.v, false, fb.v, (short)0, r0, false, false);
                r1 = __builtin_amdgcn_wmma_f32_16x16x32_bf16(
                         false, fa1.v, false, fb.v, (short)0, r1, false, false);
            }
        }
    }

    // C/D layout: vector elem i -> row (half*8 + i), column nt*16 + mrow
    #pragma unroll
    for (int rtile = 0; rtile < 2; ++rtile) {
        #pragma unroll
        for (int t = 0; t < 3; ++t) {
            v8f acc = (rtile == 0) ? ((t == 0) ? a00 : ((t == 1) ? a01 : a02))
                                   : ((t == 0) ? a10 : ((t == 1) ? a11 : a12));
            int col = wave * 48 + t * 16 + mrow;
            #pragma unroll
            for (int i = 0; i < 8; ++i) {
                int row = rt * MROWS + rtile * 16 + half * 8 + i;
                out[(long)row * CCH + col] = acc[i];
            }
        }
    }
}

// ---- per-channel sum / sumsq ----
__global__ void mk_bn_stats(const float* __restrict__ x, float* __restrict__ stats,
                            int rowsPerBlock) {
    int c = threadIdx.x;                           // 0..191 -> coalesced rows
    long r0 = (long)blockIdx.x * rowsPerBlock;
    float s = 0.f, s2 = 0.f;
    for (int r = 0; r < rowsPerBlock; ++r) {
        float v = x[(r0 + r) * CCH + c];
        s += v; s2 += v * v;
    }
    unsafeAtomicAdd(&stats[c], s);
    unsafeAtomicAdd(&stats[CCH + c], s2);
}

// ---- BN + ReLU -> bf16 (with zero pad row) ----
__global__ void mk_bn_relu_bf16(const float* __restrict__ x, const float* __restrict__ stats,
                                const float* __restrict__ gamma, const float* __restrict__ beta,
                                unsigned short* __restrict__ o) {
    long i = (long)blockIdx.x * blockDim.x + threadIdx.x;
    const long total = (long)(NPTS + 1) * CCH;
    if (i >= total) return;
    if (i >= (long)NPTS * CCH) { o[i] = 0; return; }
    int c = (int)(i % CCH);
    float mean = stats[c] * (1.f / NPTS);
    float var  = stats[CCH + c] * (1.f / NPTS) - mean * mean;
    float v = (x[i] - mean) * rsqrtf(var + 1e-5f) * gamma[c] + beta[c];
    o[i] = f2bf(v > 0.f ? v : 0.f);
}

// ---- BN + residual + ReLU -> f32 out ----
__global__ void mk_bn_add_relu(const float* __restrict__ x, const float* __restrict__ stats,
                               const float* __restrict__ gamma, const float* __restrict__ beta,
                               const float* __restrict__ res, float* __restrict__ o) {
    long i = (long)blockIdx.x * blockDim.x + threadIdx.x;
    if (i >= (long)NPTS * CCH) return;
    int c = (int)(i % CCH);
    float mean = stats[c] * (1.f / NPTS);
    float var  = stats[CCH + c] * (1.f / NPTS) - mean * mean;
    float v = (x[i] - mean) * rsqrtf(var + 1e-5f) * gamma[c] + beta[c] + res[i];
    o[i] = v > 0.f ? v : 0.f;
}

extern "C" void kernel_launch(void* const* d_in, const int* in_sizes, int n_in,
                              void* d_out, int out_size, void* d_ws, size_t ws_size,
                              hipStream_t stream) {
    const float* feats  = (const float*)d_in[0];
    const float* W1     = (const float*)d_in[1];
    const float* gamma1 = (const float*)d_in[2];
    const float* beta1  = (const float*)d_in[3];
    const float* W2     = (const float*)d_in[4];
    const float* gamma2 = (const float*)d_in[5];
    const float* beta2  = (const float*)d_in[6];
    const int*   neigh  = (const int*)d_in[7];
    float* out = (float*)d_out;

    // workspace carve-up (256B aligned)
    size_t off = 0;
    auto carve = [&](size_t bytes) {
        void* p = (char*)d_ws + off;
        off += (bytes + 255) & ~(size_t)255;
        return p;
    };
    unsigned short* featsb = (unsigned short*)carve((size_t)(NPTS + 1) * CCH * 2);
    unsigned short* midb   = (unsigned short*)carve((size_t)(NPTS + 1) * CCH * 2);
    unsigned short* w1p    = (unsigned short*)carve((size_t)KOFF * CCH * CCH * 2);
    unsigned short* w2p    = (unsigned short*)carve((size_t)KOFF * CCH * CCH * 2);
    float*          raw    = (float*)carve((size_t)NPTS * CCH * 4);
    float*          stats1 = (float*)carve(2 * CCH * 4);
    float*          stats2 = (float*)carve(2 * CCH * 4);

    hipMemsetAsync(stats1, 0, 2 * CCH * 4, stream);
    hipMemsetAsync(stats2, 0, 2 * CCH * 4, stream);

    const long totPad = (long)(NPTS + 1) * CCH;
    const long totW   = (long)KOFF * CCH * CCH;
    const long totEl  = (long)NPTS * CCH;

    mk_cvt_feats<<<(int)((totPad + 255) / 256), 256, 0, stream>>>(feats, featsb);
    mk_pack_w   <<<(int)((totW + 255) / 256), 256, 0, stream>>>(W1, w1p);
    mk_pack_w   <<<(int)((totW + 255) / 256), 256, 0, stream>>>(W2, w2p);

    // conv1 -> raw
    mk_spconv<<<NPTS / MROWS, 128, 0, stream>>>(featsb, w1p, neigh, raw);
    mk_bn_stats<<<NPTS / 512, CCH, 0, stream>>>(raw, stats1, 512);
    mk_bn_relu_bf16<<<(int)((totPad + 255) / 256), 256, 0, stream>>>(raw, stats1, gamma1, beta1, midb);

    // conv2 -> raw (reuse)
    mk_spconv<<<NPTS / MROWS, 128, 0, stream>>>(midb, w2p, neigh, raw);
    mk_bn_stats<<<NPTS / 512, CCH, 0, stream>>>(raw, stats2, 512);
    mk_bn_add_relu<<<(int)((totEl + 255) / 256), 256, 0, stream>>>(raw, stats2, gamma2, beta2, feats, out);
}
